// WindowAttention_46024869544412
// MI455X (gfx1250) — matrix operand order repaired and checked by reading, not compiled, not run on hardware
//
#include <hip/hip_runtime.h>
#include <hip/hip_bf16.h>

// ---------------------------------------------------------------------------
// Window attention for MI455X (gfx1250, wave32, WMMA).
//
// Roofline: ~155 GFLOP vs ~1.2 GB of unavoidable traffic at 23.3 TB/s ->
// bandwidth bound. Strategy: f16 intermediates (native WMMA operand, half the
// bytes), fp32 accumulation via V_WMMA_F32_16X16X32_F16 for every matmul.
//
// L2 strategy (192 MB global L2):
//   - GEMM grids put the N-split in blockIdx.x (fastest) so all column-blocks
//     sharing an A row-tile dispatch back-to-back -> A read once from HBM.
//   - Streaming intermediates (qkv 384MB, final out 256MB) use non-temporal
//     stores so they don't evict the reused x/weight lines.
//
// Workspace layout (d_ws):
//   [0, 384MB)        : qkv  f16, [b][which][head][64][32]   (scale folded in Q)
//   [384MB, 512MB)    : attn out f16, [b][n][256] (head-major channels)
//   [512MB, +128KB)   : bias f32, [h][64][64]  (rel-pos table + mask)
// ---------------------------------------------------------------------------

typedef __attribute__((ext_vector_type(16))) _Float16 v16h;
typedef __attribute__((ext_vector_type(8)))  _Float16 v8h;
typedef __attribute__((ext_vector_type(8)))  float    v8f;

#define HEADS 8
#define NTOK  64
#define HD    32
#define DIM   256
#define QKV_SCALE 0.17677669529663687f  // 1/sqrt(32)

// --- WMMA helpers (A and B in contiguous k order, wave32) --------------------

__device__ __forceinline__ v8f wmma_f16(v16h a, v16h b, v8f c) {
  return __builtin_amdgcn_wmma_f32_16x16x32_f16(false, a, false, b,
                                                (short)0, c, false, false);
}

// A fragment: 16x32 f16, row-major source with row pitch ld (halves).
// lane m = lane&15; lanes 0-15 carry K = 0..15, lanes 16-31 carry K = 16..31.
__device__ __forceinline__ v16h load_a_frag(const _Float16* p, int ld) {
  const int lane = threadIdx.x & 31;
  const int m  = lane & 15;
  const int hs = (lane >> 4) * 16;
  return *(const v16h*)(p + m * ld + hs);
}

// B fragment: 32x16 f16, loaded from the TRANSPOSED matrix Bt (N x K,
// row-major, pitch ld in halves): lane n = lane&15 reads 16 contiguous
// halves at K offset 16*(lane>>4).
__device__ __forceinline__ v16h load_bt_frag(const _Float16* p, int ld) {
  const int lane = threadIdx.x & 31;
  const int n  = lane & 15;
  const int ks = (lane >> 4) * 16;
  return *(const v16h*)(p + n * ld + ks);
}

// --- kernel 0: bias[h][i][j] = table[rel[i][j]][h] + mask[h][i][j] ----------

__global__ void wa_bias_kernel(const float* __restrict__ attn_mask,
                               const float* __restrict__ table,
                               const int* __restrict__ rel,
                               float* __restrict__ bias) {
  int t = blockIdx.x * blockDim.x + threadIdx.x;
  if (t >= HEADS * NTOK * NTOK) return;
  int h  = t >> 12;        // /4096
  int ij = t & 4095;
  bias[t] = table[rel[ij] * HEADS + h] + attn_mask[t];
}

// --- kernel 1: qkv = x @ w_qkv^T, f16 out, permuted + Q scaled --------------
// Block tile 64(M) x 128(N), 256 threads = 8 waves (2x4), 2x2 WMMA tiles/wave.
// blockIdx.x = N split (fastest -> L2 reuse of the x tile), blockIdx.y = M.

__global__ __launch_bounds__(256)
void wa_qkv_gemm(const float* __restrict__ x,
                 const float* __restrict__ w_qkv,
                 _Float16* __restrict__ qkv) {
  __shared__ __align__(32) _Float16 Alds[64 * 32];
  __shared__ __align__(32) _Float16 Btlds[128 * 32];

  const int tid   = threadIdx.x;
  const int wave  = tid >> 5;
  const int lane  = tid & 31;
  const int n15   = lane & 15;
  const int hi    = lane >> 4;
  const int waveM = wave >> 2;           // 0..1
  const int waveN = wave & 3;            // 0..3
  const size_t rowBase = (size_t)blockIdx.y * 64;
  const int jBase = blockIdx.x * 128;

  v8f acc[2][2] = {};

  for (int k0 = 0; k0 < DIM; k0 += 32) {
    // stage A: 64x32 f32 -> f16 (8 elems / thread, contiguous)
    {
      const int idx = tid * 8;
      const int r = idx >> 5, c = idx & 31;
      const float4* s4 = (const float4*)(x + (rowBase + r) * DIM + k0 + c);
      float4 f0 = s4[0], f1 = s4[1];
      _Float16* dst = &Alds[r * 32 + c];
      dst[0] = (_Float16)f0.x; dst[1] = (_Float16)f0.y;
      dst[2] = (_Float16)f0.z; dst[3] = (_Float16)f0.w;
      dst[4] = (_Float16)f1.x; dst[5] = (_Float16)f1.y;
      dst[6] = (_Float16)f1.z; dst[7] = (_Float16)f1.w;
      if (k0 + 32 < DIM) __builtin_prefetch(s4 + 8, 0, 1); // next K tile
    }
    // stage Bt: rows of w_qkv (128x32), f32 -> f16 (16 elems / thread)
    {
      const int idx = tid * 16;
      const int r = idx >> 5, c = idx & 31;          // c in {0,16}
      const float4* s4 = (const float4*)(w_qkv + (size_t)(jBase + r) * DIM + k0 + c);
      _Float16* dst = &Btlds[r * 32 + c];
#pragma unroll
      for (int q = 0; q < 4; ++q) {
        float4 f = s4[q];
        dst[q * 4 + 0] = (_Float16)f.x; dst[q * 4 + 1] = (_Float16)f.y;
        dst[q * 4 + 2] = (_Float16)f.z; dst[q * 4 + 3] = (_Float16)f.w;
      }
    }
    __syncthreads();

    v16h a0 = load_a_frag(&Alds[(waveM * 32) * 32], 32);
    v16h a1 = load_a_frag(&Alds[(waveM * 32 + 16) * 32], 32);
    v16h b0 = load_bt_frag(&Btlds[(waveN * 32) * 32], 32);
    v16h b1 = load_bt_frag(&Btlds[(waveN * 32 + 16) * 32], 32);
    acc[0][0] = wmma_f16(a0, b0, acc[0][0]);
    acc[0][1] = wmma_f16(a0, b1, acc[0][1]);
    acc[1][0] = wmma_f16(a1, b0, acc[1][0]);
    acc[1][1] = wmma_f16(a1, b1, acc[1][1]);
    __syncthreads();
  }

  // scatter-store into [b][which][head][n][d] f16; fold softmax scale into Q.
  // Non-temporal: qkv is a 384MB stream, keep it out of the hot L2 set.
  const size_t b = blockIdx.y;   // rowBase is 64-aligned -> one window / block
#pragma unroll
  for (int ti = 0; ti < 2; ++ti) {
#pragma unroll
    for (int tj = 0; tj < 2; ++tj) {
      const int j = jBase + waveN * 32 + tj * 16 + n15;
      const int which = j >> 8;
      const int head  = (j >> 5) & 7;
      const int dd    = j & 31;
      const float mul = (which == 0) ? QKV_SCALE : 1.0f;
      _Float16* base = qkv + ((b * 3 + which) * HEADS + head) * (NTOK * HD) + dd;
#pragma unroll
      for (int r = 0; r < 8; ++r) {
        const int n = waveM * 32 + ti * 16 + r + 8 * hi;
        __builtin_nontemporal_store((_Float16)(acc[ti][tj][r] * mul),
                                    base + n * HD);
      }
    }
  }
}

// --- kernel 2: per (window, head) attention ---------------------------------
// 128 threads = 4 waves; wave w owns rows [16w, 16w+16).
// S = Q K^T (16 WMMAs) -> LDS; softmax fp32; O = P V (16 WMMAs).

__global__ __launch_bounds__(128)
void wa_attn(const _Float16* __restrict__ qkv,
             const float* __restrict__ bias,
             _Float16* __restrict__ outw) {
  __shared__ __align__(32) float    S[NTOK * NTOK];    // 16 KB
  __shared__ __align__(32) _Float16 P[NTOK * NTOK];    //  8 KB
  __shared__ __align__(32) _Float16 Vt[HD * NTOK];     //  4 KB

  const int bh = blockIdx.x;
  const int b = bh >> 3, h = bh & 7;
  const _Float16* Q = qkv + (((size_t)b * 3 + 0) * HEADS + h) * (NTOK * HD);
  const _Float16* K = qkv + (((size_t)b * 3 + 1) * HEADS + h) * (NTOK * HD);
  const _Float16* V = qkv + (((size_t)b * 3 + 2) * HEADS + h) * (NTOK * HD);

  const int tid  = threadIdx.x;
  const int wave = tid >> 5;
  const int lane = tid & 31;
  const int n15  = lane & 15;
  const int hi   = lane >> 4;

  // stage V^T (HD x NTOK) so PV B-fragments are contiguous row loads
  {
    const int j  = tid >> 1;            // token 0..63
    const int d0 = (tid & 1) * 16;
#pragma unroll
    for (int i = 0; i < 16; ++i) Vt[(d0 + i) * NTOK + j] = V[j * HD + d0 + i];
  }

  // S = (scaled Q) K^T : K acts as Bt (row-major token x d)
  const int m0 = wave * 16;
  const v16h aq = load_a_frag(Q + m0 * HD, HD);
#pragma unroll
  for (int t = 0; t < 4; ++t) {
    v16h bk = load_bt_frag(K + (t * 16) * HD, HD);
    v8f s = {};
    s = wmma_f16(aq, bk, s);
#pragma unroll
    for (int r = 0; r < 8; ++r)
      S[(m0 + r + 8 * hi) * NTOK + t * 16 + n15] = s[r];
  }
  __syncthreads();

  // softmax over j for each row i (threads 0..63), bias added pre-softmax
  if (tid < NTOK) {
    const float* bp = bias + h * (NTOK * NTOK) + tid * NTOK;
    float* sp = &S[tid * NTOK];
    float mx = -3.0e38f;
#pragma unroll 8
    for (int j = 0; j < NTOK; ++j) {
      float v = sp[j] + bp[j];
      sp[j] = v;
      mx = fmaxf(mx, v);
    }
    float sum = 0.0f;
#pragma unroll 8
    for (int j = 0; j < NTOK; ++j) {
      float e = __expf(sp[j] - mx);
      sp[j] = e;
      sum += e;
    }
    const float inv = 1.0f / sum;
#pragma unroll 8
    for (int j = 0; j < NTOK; ++j) P[tid * NTOK + j] = (_Float16)(sp[j] * inv);
  }
  __syncthreads();

  // O = P V : two 16x16 output tiles per wave, K = 64 in two 32-steps
  v8f o[2] = {};
#pragma unroll
  for (int k0 = 0; k0 < NTOK; k0 += 32) {
    v16h ap  = load_a_frag(&P[m0 * NTOK + k0], NTOK);
    v16h bv0 = load_bt_frag(&Vt[0 * NTOK + k0], NTOK);
    v16h bv1 = load_bt_frag(&Vt[16 * NTOK + k0], NTOK);
    o[0] = wmma_f16(ap, bv0, o[0]);
    o[1] = wmma_f16(ap, bv1, o[1]);
  }
#pragma unroll
  for (int t = 0; t < 2; ++t) {
#pragma unroll
    for (int r = 0; r < 8; ++r) {
      const int n = m0 + r + 8 * hi;
      outw[((size_t)b * NTOK + n) * DIM + h * HD + t * 16 + n15] = (_Float16)o[t][r];
    }
  }
}

// --- kernel 3: out = attn_f16 @ w_proj^T + b_proj (fp32 out) ----------------
// Same tiling as kernel 1; A is already f16. blockIdx.x = N split (L2 reuse).

__global__ __launch_bounds__(256)
void wa_proj_gemm(const _Float16* __restrict__ ain,
                  const float* __restrict__ w_proj,
                  const float* __restrict__ b_proj,
                  float* __restrict__ out) {
  __shared__ __align__(32) _Float16 Alds[64 * 32];
  __shared__ __align__(32) _Float16 Btlds[128 * 32];

  const int tid   = threadIdx.x;
  const int wave  = tid >> 5;
  const int lane  = tid & 31;
  const int n15   = lane & 15;
  const int hi    = lane >> 4;
  const int waveM = wave >> 2;
  const int waveN = wave & 3;
  const size_t rowBase = (size_t)blockIdx.y * 64;
  const int jBase = blockIdx.x * 128;

  v8f acc[2][2] = {};

  for (int k0 = 0; k0 < DIM; k0 += 32) {
    {
      const int idx = tid * 8;
      const int r = idx >> 5, c = idx & 31;
      const _Float16* src = ain + (rowBase + r) * DIM + k0 + c;
      *(v8h*)&Alds[r * 32 + c] = *(const v8h*)src;
      if (k0 + 32 < DIM) __builtin_prefetch(src + 32, 0, 1);
    }
    {
      const int idx = tid * 16;
      const int r = idx >> 5, c = idx & 31;
      const float4* s4 = (const float4*)(w_proj + (size_t)(jBase + r) * DIM + k0 + c);
      _Float16* dst = &Btlds[r * 32 + c];
#pragma unroll
      for (int q = 0; q < 4; ++q) {
        float4 f = s4[q];
        dst[q * 4 + 0] = (_Float16)f.x; dst[q * 4 + 1] = (_Float16)f.y;
        dst[q * 4 + 2] = (_Float16)f.z; dst[q * 4 + 3] = (_Float16)f.w;
      }
    }
    __syncthreads();

    v16h a0 = load_a_frag(&Alds[(waveM * 32) * 32], 32);
    v16h a1 = load_a_frag(&Alds[(waveM * 32 + 16) * 32], 32);
    v16h b0 = load_bt_frag(&Btlds[(waveN * 32) * 32], 32);
    v16h b1 = load_bt_frag(&Btlds[(waveN * 32 + 16) * 32], 32);
    acc[0][0] = wmma_f16(a0, b0, acc[0][0]);
    acc[0][1] = wmma_f16(a0, b1, acc[0][1]);
    acc[1][0] = wmma_f16(a1, b0, acc[1][0]);
    acc[1][1] = wmma_f16(a1, b1, acc[1][1]);
    __syncthreads();
  }

  // fp32 result + bias; final output is write-once -> non-temporal stores.
#pragma unroll
  for (int ti = 0; ti < 2; ++ti) {
#pragma unroll
    for (int tj = 0; tj < 2; ++tj) {
      const int j = jBase + waveN * 32 + tj * 16 + n15;
      const float bj = b_proj[j];
#pragma unroll
      for (int r = 0; r < 8; ++r) {
        const size_t m = rowBase + waveM * 32 + ti * 16 + r + 8 * hi;
        __builtin_nontemporal_store(acc[ti][tj][r] + bj, out + m * DIM + j);
      }
    }
  }
}

// ---------------------------------------------------------------------------

extern "C" void kernel_launch(void* const* d_in, const int* in_sizes, int n_in,
                              void* d_out, int out_size, void* d_ws, size_t ws_size,
                              hipStream_t stream) {
  const float* x         = (const float*)d_in[0];
  const float* attn_mask = (const float*)d_in[1];
  const float* w_qkv     = (const float*)d_in[2];
  const float* w_proj    = (const float*)d_in[3];
  const float* b_proj    = (const float*)d_in[4];
  const float* table     = (const float*)d_in[5];
  const int*   rel       = (const int*)d_in[6];
  float* out = (float*)d_out;

  char* ws = (char*)d_ws;
  _Float16* qkv  = (_Float16*)ws;                          // 384 MB
  _Float16* aout = (_Float16*)(ws + 402653184ull);         // 128 MB
  float*    bias = (float*)(ws + 536870912ull);            // 128 KB

  wa_bias_kernel<<<128, 256, 0, stream>>>(attn_mask, table, rel, bias);
  // N-split in blockIdx.x so column-blocks sharing an A tile run adjacently.
  wa_qkv_gemm<<<dim3(6, 4096), 256, 0, stream>>>(x, w_qkv, qkv);
  wa_attn<<<4096 * HEADS, 128, 0, stream>>>(qkv, bias, aout);
  wa_proj_gemm<<<dim3(2, 4096), 256, 0, stream>>>(aout, w_proj, b_proj, out);
}
